// MultiHeadAttention_78838419685675
// MI455X (gfx1250) — compile-verified, matrix operand order checked by reading, not run on hardware
//
#include <hip/hip_runtime.h>
#include <hip/hip_bf16.h>

// ---------------- problem constants ----------------
#define HIDW 2048
#define NH   16
#define NKV  2
#define HD   128
#define BB   2
#define SS   1024
#define MROWS (BB*SS)   // 2048
#define QN   (NH*HD)    // 2048
#define KVN  (NKV*HD)   // 256

typedef __attribute__((ext_vector_type(16))) __bf16 v16bf;
typedef __attribute__((ext_vector_type(8)))  __bf16 v8bf;
typedef __attribute__((ext_vector_type(8)))  float  v8f;

union ABFrag { v16bf v; struct { v8bf lo, hi; } h; };

__device__ __forceinline__ __bf16 f2bf(float f) {
  unsigned u = __builtin_bit_cast(unsigned, f);
  u += 0x7FFFu + ((u >> 16) & 1u);               // round-to-nearest-even
  unsigned short hsv = (unsigned short)(u >> 16);
  return __builtin_bit_cast(__bf16, hsv);
}

// ---------------- fp32 -> bf16 convert ----------------
__global__ void mha_f32_to_bf16(const float* __restrict__ in, __bf16* __restrict__ outp, int n) {
  int i = blockIdx.x * blockDim.x + threadIdx.x;
  int stride = gridDim.x * blockDim.x;
  for (; i < n; i += stride) outp[i] = f2bf(in[i]);
}

// ---------------- fp32 (K x N) -> bf16 transposed (N x K) ----------------
__global__ void mha_transpose_bf16(const float* __restrict__ W, __bf16* __restrict__ Wt,
                                   int K, int N) {
  __shared__ float tile[32][33];
  int n0 = blockIdx.x * 32, k0 = blockIdx.y * 32;
#pragma unroll
  for (int i = 0; i < 32; i += 8)
    tile[threadIdx.y + i][threadIdx.x] =
        W[(size_t)(k0 + threadIdx.y + i) * N + n0 + threadIdx.x];
  __syncthreads();
#pragma unroll
  for (int i = 0; i < 32; i += 8)
    Wt[(size_t)(n0 + threadIdx.y + i) * K + k0 + threadIdx.x] =
        f2bf(tile[threadIdx.x][threadIdx.y + i]);
}

// ---------------- bf16 WMMA GEMM: C(f32, MxN) = A(MxK) * Bt(NxK)^T + bias ----------------
// block = 8 waves; block tile = 32 (M) x 512 (N); wave tile = 32 x 64.
// A panel (32 x 32 per K-step) staged through LDS with async-to-LDS, double buffered.
__global__ void mha_gemm_bf16(const __bf16* __restrict__ A, const __bf16* __restrict__ Bt,
                              const float* __restrict__ bias, float* __restrict__ C,
                              int M, int N, int K) {
  __shared__ __align__(16) __bf16 sA[2][32 * 32];   // 2 x 2KB
  const int lane = threadIdx.x & 31;
  const int wv   = threadIdx.x >> 5;
  const int mt   = blockIdx.y;                      // 32-row tile index
  const int n0raw = (blockIdx.x * 8 + wv) * 64;
  const bool active = n0raw < N;
  const int n0 = active ? n0raw : 0;                // clamp so dead waves still load valid mem
  const int rc   = lane & 15;
  const int half = lane >> 4;

  // async loader geometry: waves 0..3, each lane copies one 16B chunk (2KB total / step)
  const int ldRow = (wv & 3) * 8 + (lane >> 2);     // 0..31
  const int ldCol = (lane & 3) * 8;                 // 0,8,16,24 (elements)
  const __bf16* ldG = A + (size_t)(mt * 32 + ldRow) * K + ldCol;
  const unsigned ldL0 = (unsigned)(uintptr_t)(&sA[0][ldRow * 32 + ldCol]);
  const unsigned ldL1 = (unsigned)(uintptr_t)(&sA[1][ldRow * 32 + ldCol]);

  if (wv < 4) {                                     // prologue: fill buffer 0 (k0 = 0)
    asm volatile("global_load_async_to_lds_b128 %0, %1, off"
                 :: "v"(ldL0), "v"((unsigned long long)(uintptr_t)ldG) : "memory");
  }

  const __bf16* Brow0 = Bt + (size_t)(n0 + rc) * K;
  v8f acc[8] = {};
  for (int k0 = 0; k0 < K; k0 += 32) {
    asm volatile("s_wait_asynccnt 0x0" ::: "memory");
    __syncthreads();                                // A panel for this step visible to all waves
    const int cur = (k0 >> 5) & 1;
    if (wv < 4 && (k0 + 32) < K) {                  // issue next panel into the other buffer
      unsigned dst = cur ? ldL0 : ldL1;
      asm volatile("global_load_async_to_lds_b128 %0, %1, off"
                   :: "v"(dst),
                      "v"((unsigned long long)(uintptr_t)(ldG + k0 + 32)) : "memory");
    }
    __builtin_prefetch(Brow0 + k0 + 256, 0, 1);
    const __bf16* sAc = &sA[cur][0];
    ABFrag a0, a1;
    a0.h.lo = *(const v8bf*)(sAc + rc * 32 + half * 8);
    a0.h.hi = *(const v8bf*)(sAc + rc * 32 + 16 + half * 8);
    a1.h.lo = *(const v8bf*)(sAc + (16 + rc) * 32 + half * 8);
    a1.h.hi = *(const v8bf*)(sAc + (16 + rc) * 32 + 16 + half * 8);
#pragma unroll
    for (int t = 0; t < 4; ++t) {
      const __bf16* Br = Brow0 + (size_t)t * 16 * K;
      ABFrag b;
      b.h.lo = *(const v8bf*)(Br + k0 + half * 8);
      b.h.hi = *(const v8bf*)(Br + k0 + 16 + half * 8);
      acc[t] = __builtin_amdgcn_wmma_f32_16x16x32_bf16(
          false, a0.v, false, b.v, (short)0, acc[t], false, false);
      acc[4 + t] = __builtin_amdgcn_wmma_f32_16x16x32_bf16(
          false, a1.v, false, b.v, (short)0, acc[4 + t], false, false);
    }
  }
  if (active) {
#pragma unroll
    for (int t = 0; t < 4; ++t) {
      float bv = bias ? bias[n0 + t * 16 + rc] : 0.0f;
#pragma unroll
      for (int r = 0; r < 8; ++r) {
        int m0 = mt * 32 + half * 8 + r;            // first 16-row tile
        C[(size_t)m0 * N + n0 + t * 16 + rc] = acc[t][r] + bv;
        C[(size_t)(m0 + 16) * N + n0 + t * 16 + rc] = acc[4 + t][r] + bv;
      }
    }
  }
}

// ---------------- QKV epilogue: RMS-norm + sectioned RoPE + mask, fp32 -> bf16 ----------------
// one wave per (b, s, head) row of 128 values (4 per lane).
__global__ void mha_qkv_epilogue(const float* __restrict__ Qraw, const float* __restrict__ Kraw,
                                 const float* __restrict__ Vraw, const float* __restrict__ qg,
                                 const float* __restrict__ kg, const float* __restrict__ cosb,
                                 const float* __restrict__ sinb, const float* __restrict__ mask,
                                 __bf16* __restrict__ Qh, __bf16* __restrict__ Kh,
                                 __bf16* __restrict__ Vt) {
  const int lane = threadIdx.x & 31;
  const int wid  = blockIdx.x * (blockDim.x >> 5) + (threadIdx.x >> 5);
  const int NQ   = MROWS * NH;    // 32768 Q rows
  const int NKVR = MROWS * NKV;   // 4096 K rows (then 4096 V rows)

  const float* src; const float* gamma = nullptr;
  int b, s, mode, headQ = 0, headKV = 0;
  if (wid < NQ) {
    mode = 0; int m = wid >> 4; headQ = wid & 15; b = m >> 10; s = m & 1023;
    src = Qraw + (size_t)m * QN + headQ * HD; gamma = qg;
  } else if (wid < NQ + NKVR) {
    int r2 = wid - NQ; mode = 1; int m = r2 >> 1; headKV = r2 & 1; b = m >> 10; s = m & 1023;
    src = Kraw + (size_t)m * KVN + headKV * HD; gamma = kg;
  } else {
    int r3 = wid - NQ - NKVR; mode = 2; int m = r3 >> 1; headKV = r3 & 1; b = m >> 10; s = m & 1023;
    src = Vraw + (size_t)m * KVN + headKV * HD;
  }

  float val[4];
#pragma unroll
  for (int j = 0; j < 4; ++j) val[j] = src[lane + 32 * j];
  const float mval = mask[b * SS + s];

  if (mode == 2) {   // V: mask only, store transposed [HD][S]
    __bf16* dst = Vt + ((size_t)(b * NKV + headKV) * HD) * SS + s;
#pragma unroll
    for (int j = 0; j < 4; ++j) dst[(size_t)(lane + 32 * j) * SS] = f2bf(val[j] * mval);
    return;
  }

  // RMS norm over 128 elems
  float sq = 0.f;
#pragma unroll
  for (int j = 0; j < 4; ++j) sq += val[j] * val[j];
  sq += __shfl_xor(sq, 1, 32);  sq += __shfl_xor(sq, 2, 32);
  sq += __shfl_xor(sq, 4, 32);  sq += __shfl_xor(sq, 8, 32);
  sq += __shfl_xor(sq, 16, 32);
  const float inv = rsqrtf(sq * (1.0f / HD) + 1e-6f);

  float nv[4];
#pragma unroll
  for (int j = 0; j < 4; ++j) {
    int d = lane + 32 * j;
    nv[j] = val[j] * inv * gamma[d];
  }
  // sectioned RoPE:  sel(d) = (d&63)<16 ? 0 : (d&63)<40 ? 1 : 2
  float ov[4];
#pragma unroll
  for (int j = 0; j < 4; ++j) {
    int d  = lane + 32 * j;
    int dd = d & 63;
    int sel = dd < 16 ? 0 : (dd < 40 ? 1 : 2);
    size_t cidx = (((size_t)sel * BB + b) * SS + s) * HD + d;
    float c = cosb[cidx], sn = sinb[cidx];
    int pj = (j < 2) ? j + 2 : j - 2;           // rotate_half partner (d +/- 64 in same lane)
    float rh = (j < 2) ? -nv[pj] : nv[pj];
    ov[j] = nv[j] * c + rh * sn;
  }
  if (mode == 0) {
    __bf16* dst = Qh + ((size_t)(b * NH + headQ) * SS + s) * HD;
#pragma unroll
    for (int j = 0; j < 4; ++j) dst[lane + 32 * j] = f2bf(ov[j]);
  } else {
    __bf16* dst = Kh + ((size_t)(b * NKV + headKV) * SS + s) * HD;
#pragma unroll
    for (int j = 0; j < 4; ++j) dst[lane + 32 * j] = f2bf(ov[j] * mval);
  }
}

// ---------------- flash attention: one wave per 16-query tile ----------------
__global__ void mha_flash_attn(const __bf16* __restrict__ Qh, const __bf16* __restrict__ Kh,
                               const __bf16* __restrict__ Vt, const float* __restrict__ mask,
                               __bf16* __restrict__ Oattn) {
  __shared__ __align__(16) __bf16 Pld[8][16 * 32];
  const int lane = threadIdx.x & 31;
  const int wv   = threadIdx.x >> 5;
  const int wid  = blockIdx.x * 8 + wv;        // 2048 query tiles
  const int qt = wid & 63;
  const int bh = wid >> 6;
  const int b  = bh >> 4;
  const int h  = bh & 15;
  const int kv = h >> 3;                        // GQA group (G=8)
  const int rc   = lane & 15;
  const int half = lane >> 4;

  const __bf16* Qbase = Qh + ((size_t)(b * NH + h) * SS + qt * 16) * HD;
  const __bf16* Kbase = Kh + ((size_t)(b * NKV + kv) * SS) * HD;
  const __bf16* Vbase = Vt + ((size_t)(b * NKV + kv) * HD) * SS;
  const float*  mrow  = mask + b * SS;
  __bf16* pbuf = &Pld[wv][0];

  v8f oacc[8] = {};
  float mst[8], lst[8];
#pragma unroll
  for (int r = 0; r < 8; ++r) { mst[r] = -1e30f; lst[r] = 0.f; }

  const float scale = 0.08838834764831843f;     // HD^-0.5
  const int nkb = qt / 2 + 1;                   // causal 32-key blocks

  for (int kb = 0; kb < nkb; ++kb) {
    const int k0 = kb * 32;
    v8f s0 = {}, s1 = {};
    const __bf16* Arow = Qbase + rc * HD;
    const __bf16* B0   = Kbase + (size_t)(k0 + rc) * HD;
    const __bf16* B1   = B0 + (size_t)16 * HD;
#pragma unroll
    for (int hd0 = 0; hd0 < HD; hd0 += 32) {
      ABFrag a, b0, b1;
      a.h.lo  = *(const v8bf*)(Arow + hd0 + half * 8);
      a.h.hi  = *(const v8bf*)(Arow + hd0 + 16 + half * 8);
      b0.h.lo = *(const v8bf*)(B0 + hd0 + half * 8);
      b0.h.hi = *(const v8bf*)(B0 + hd0 + 16 + half * 8);
      b1.h.lo = *(const v8bf*)(B1 + hd0 + half * 8);
      b1.h.hi = *(const v8bf*)(B1 + hd0 + 16 + half * 8);
      s0 = __builtin_amdgcn_wmma_f32_16x16x32_bf16(false, a.v, false, b0.v, (short)0, s0, false, false);
      s1 = __builtin_amdgcn_wmma_f32_16x16x32_bf16(false, a.v, false, b1.v, (short)0, s1, false, false);
    }
    // mask + online softmax
    const float hb0 = (1.0f - mrow[k0 + rc])      * -1e9f;
    const float hb1 = (1.0f - mrow[k0 + 16 + rc]) * -1e9f;
    float p0[8], p1[8], alpha[8];
#pragma unroll
    for (int r = 0; r < 8; ++r) {
      int qgl = qt * 16 + half * 8 + r;
      float v0 = s0[r] * scale + hb0 + ((k0 + rc)      <= qgl ? 0.f : -1e9f);
      float v1 = s1[r] * scale + hb1 + ((k0 + 16 + rc) <= qgl ? 0.f : -1e9f);
      float t = fmaxf(v0, v1);
      t = fmaxf(t, __shfl_xor(t, 1, 32));
      t = fmaxf(t, __shfl_xor(t, 2, 32));
      t = fmaxf(t, __shfl_xor(t, 4, 32));
      t = fmaxf(t, __shfl_xor(t, 8, 32));
      float mnew = fmaxf(mst[r], t);
      alpha[r] = __expf(mst[r] - mnew);
      mst[r] = mnew;
      p0[r] = __expf(v0 - mnew);
      p1[r] = __expf(v1 - mnew);
      float rs = p0[r] + p1[r];
      rs += __shfl_xor(rs, 1, 32);
      rs += __shfl_xor(rs, 2, 32);
      rs += __shfl_xor(rs, 4, 32);
      rs += __shfl_xor(rs, 8, 32);
      lst[r] = lst[r] * alpha[r] + rs;
    }
#pragma unroll
    for (int t = 0; t < 8; ++t)
#pragma unroll
      for (int r = 0; r < 8; ++r) oacc[t][r] *= alpha[r];

    // stage P (16x32) through LDS to get A-fragment layout
#pragma unroll
    for (int r = 0; r < 8; ++r) {
      int m = half * 8 + r;
      pbuf[m * 32 + rc]      = f2bf(p0[r]);
      pbuf[m * 32 + 16 + rc] = f2bf(p1[r]);
    }
    asm volatile("s_wait_dscnt 0" ::: "memory");
    ABFrag pa;
    pa.h.lo = *(const v8bf*)(pbuf + rc * 32 + half * 8);
    pa.h.hi = *(const v8bf*)(pbuf + rc * 32 + 16 + half * 8);
#pragma unroll
    for (int t = 0; t < 8; ++t) {
      const __bf16* Vrow = Vbase + (size_t)(t * 16 + rc) * SS + k0;
      ABFrag bv;
      bv.h.lo = *(const v8bf*)(Vrow + half * 8);
      bv.h.hi = *(const v8bf*)(Vrow + 16 + half * 8);
      oacc[t] = __builtin_amdgcn_wmma_f32_16x16x32_bf16(
          false, pa.v, false, bv.v, (short)0, oacc[t], false, false);
    }
    asm volatile("s_wait_dscnt 0" ::: "memory");
  }

  // write normalized output, layout [b][s][h*HD]
  const size_t obase = ((size_t)(b * SS) + qt * 16) * (size_t)QN + h * HD;
#pragma unroll
  for (int r = 0; r < 8; ++r) {
    float invl = 1.0f / lst[r];
    int m = half * 8 + r;
#pragma unroll
    for (int t = 0; t < 8; ++t)
      Oattn[obase + (size_t)m * QN + t * 16 + rc] = f2bf(oacc[t][r] * invl);
  }
}

// ---------------- host launcher ----------------
extern "C" void kernel_launch(void* const* d_in, const int* in_sizes, int n_in,
                              void* d_out, int out_size, void* d_ws, size_t ws_size,
                              hipStream_t stream) {
  const float* x    = (const float*)d_in[0];
  const float* cosb = (const float*)d_in[1];
  const float* sinb = (const float*)d_in[2];
  const float* mask = (const float*)d_in[3];
  const float* q_w  = (const float*)d_in[4];
  const float* q_b  = (const float*)d_in[5];
  const float* k_w  = (const float*)d_in[6];
  const float* k_b  = (const float*)d_in[7];
  const float* v_w  = (const float*)d_in[8];
  const float* v_b  = (const float*)d_in[9];
  const float* q_g  = (const float*)d_in[10];
  const float* k_g  = (const float*)d_in[11];
  const float* o_w  = (const float*)d_in[12];
  float* out = (float*)d_out;

  char* ws = (char*)d_ws;
  size_t off = 0;
  auto take = [&](size_t bytes) -> void* {
    void* p = ws + off;
    off += (bytes + 255) & ~(size_t)255;
    return p;
  };
  __bf16* x_bf = (__bf16*)take((size_t)MROWS * HIDW * 2);
  __bf16* qwT  = (__bf16*)take((size_t)QN * HIDW * 2);
  __bf16* kwT  = (__bf16*)take((size_t)KVN * HIDW * 2);
  __bf16* vwT  = (__bf16*)take((size_t)KVN * HIDW * 2);
  __bf16* owT  = (__bf16*)take((size_t)HIDW * QN * 2);
  float*  Qraw = (float*) take((size_t)MROWS * QN * 4);
  float*  Kraw = (float*) take((size_t)MROWS * KVN * 4);
  float*  Vraw = (float*) take((size_t)MROWS * KVN * 4);
  __bf16* Qh   = (__bf16*)take((size_t)MROWS * QN * 2);
  __bf16* Kh   = (__bf16*)take((size_t)MROWS * KVN * 2);
  __bf16* Vt   = (__bf16*)take((size_t)MROWS * KVN * 2);
  __bf16* attn = (__bf16*)Qraw;   // reuse Qraw region: free after epilogue, before flash writes

  // 1) convert x to bf16
  mha_f32_to_bf16<<<2048, 256, 0, stream>>>(x, x_bf, MROWS * HIDW);
  // 2) transpose+convert weights to (N x K) bf16
  mha_transpose_bf16<<<dim3(QN / 32, HIDW / 32),  dim3(32, 8), 0, stream>>>(q_w, qwT, HIDW, QN);
  mha_transpose_bf16<<<dim3(KVN / 32, HIDW / 32), dim3(32, 8), 0, stream>>>(k_w, kwT, HIDW, KVN);
  mha_transpose_bf16<<<dim3(KVN / 32, HIDW / 32), dim3(32, 8), 0, stream>>>(v_w, vwT, HIDW, KVN);
  mha_transpose_bf16<<<dim3(QN / 32, HIDW / 32),  dim3(32, 8), 0, stream>>>(o_w, owT, HIDW, QN);
  // 3) QKV projections (WMMA, 32-row M tiles)
  mha_gemm_bf16<<<dim3(4, MROWS / 32), 256, 0, stream>>>(x_bf, qwT, q_b, Qraw, MROWS, QN,  HIDW);
  mha_gemm_bf16<<<dim3(1, MROWS / 32), 256, 0, stream>>>(x_bf, kwT, k_b, Kraw, MROWS, KVN, HIDW);
  mha_gemm_bf16<<<dim3(1, MROWS / 32), 256, 0, stream>>>(x_bf, vwT, v_b, Vraw, MROWS, KVN, HIDW);
  // 4) RMS-norm + RoPE + mask epilogue
  mha_qkv_epilogue<<<5120, 256, 0, stream>>>(Qraw, Kraw, Vraw, q_g, k_g, cosb, sinb, mask,
                                             Qh, Kh, Vt);
  // 5) causal GQA flash attention (WMMA)
  mha_flash_attn<<<256, 256, 0, stream>>>(Qh, Kh, Vt, mask, attn);
  // 6) output projection (WMMA) -> fp32 d_out
  mha_gemm_bf16<<<dim3(4, MROWS / 32), 256, 0, stream>>>(attn, owT, nullptr, out,
                                                         MROWS, HIDW, QN);
}
